// Mamba4CTR_1073741824697
// MI455X (gfx1250) — matrix operand order repaired and checked by reading, not compiled, not run on hardware
//
#include <hip/hip_runtime.h>
#include <hip/hip_bf16.h>
#include <stdint.h>

#define NF     39
#define SEQLEN 40
#define EDIM   128
#define STDIM  16
#define NLAYER 4
#define BATCHN 2048
#define VOCABp1 10001

typedef __attribute__((ext_vector_type(16))) __bf16 v16bf;
typedef __attribute__((ext_vector_type(8)))  float  v8f;

// ---------- bf16 helpers (round-to-nearest-even) ----------
static __device__ __forceinline__ unsigned short f2bf(float f) {
  union { float f; unsigned u; } v; v.f = f;
  unsigned r = v.u + 0x7FFFu + ((v.u >> 16) & 1u);
  return (unsigned short)(r >> 16);
}
static __device__ __forceinline__ float bf2f(unsigned short h) {
  union { float f; unsigned u; } v; v.u = ((unsigned)h) << 16; return v.f;
}

// ---------- CDNA5 async global->LDS copy (ASYNCcnt-tracked, no VGPR bounce) --
// LDS operand is the low 32 bits of the flat shared address (ISA 10.2: LDS
// aperture addresses truncate to the LDS byte offset).
static __device__ __forceinline__ void async_copy_b128(void* lds, const void* gptr) {
  unsigned ldsOff = (unsigned)(size_t)lds;
  asm volatile("global_load_async_to_lds_b128 %0, %1, off"
               :: "v"(ldsOff), "v"(gptr) : "memory");
}
static __device__ __forceinline__ void wait_async0() {
  asm volatile("s_wait_asynccnt 0" ::: "memory");
}

// ---------------------------------------------------------------------------
// Weight prep: (K,N) f32 row-major -> (N,K) bf16 (column-major for the GEMM)
// ---------------------------------------------------------------------------
__global__ void __launch_bounds__(256)
wprep_kernel(const float* __restrict__ in, unsigned short* __restrict__ out,
             int K, int N)
{
  const int idx = blockIdx.x * 256 + threadIdx.x;
  if (idx >= K * N) return;
  const int k = idx / N, n = idx % N;
  out[(size_t)n * K + k] = f2bf(in[idx]);
}

// ---------------------------------------------------------------------------
// Embedding gather + cls token  ->  bf16 seq (B, 40, 128)
// ---------------------------------------------------------------------------
__global__ void __launch_bounds__(256)
embed_kernel(const int* __restrict__ x, const float* __restrict__ emb,
             const float* __restrict__ cls, unsigned short* __restrict__ seq)
{
  size_t idx = (size_t)blockIdx.x * 256 + threadIdx.x;
  const size_t total = (size_t)BATCHN * SEQLEN * EDIM;
  if (idx >= total) return;
  const int e   = (int)(idx & (EDIM - 1));
  const int tok = (int)((idx >> 7) % SEQLEN);
  const int b   = (int)(idx / ((size_t)SEQLEN * EDIM));
  float val;
  if (tok == 0) {
    val = cls[e];
  } else {
    const int f  = tok - 1;
    const int id = x[b * NF + f];
    val = (id == 0) ? 0.0f : emb[((size_t)f * VOCABp1 + id) * EDIM + e];
  }
  seq[idx] = f2bf(val);
}

// ---------------------------------------------------------------------------
// LayerNorm (per token), emits both direction-0 and direction-1 gamma/beta
// one wave32 per token, shuffle reductions
// ---------------------------------------------------------------------------
__global__ void __launch_bounds__(256)
ln_kernel(const unsigned short* __restrict__ seq,
          const float* __restrict__ g, const float* __restrict__ bta,
          unsigned short* __restrict__ xnf, unsigned short* __restrict__ xnb)
{
  const int token = blockIdx.x * 8 + (threadIdx.x >> 5);
  const int lane  = threadIdx.x & 31;
  const unsigned short* p = seq + (size_t)token * EDIM;
  float v[4];
#pragma unroll
  for (int i = 0; i < 4; ++i) v[i] = bf2f(p[lane + 32 * i]);
  float s = v[0] + v[1] + v[2] + v[3];
#pragma unroll
  for (int off = 16; off > 0; off >>= 1) s += __shfl_xor(s, off, 32);
  const float mean = s * (1.0f / EDIM);
  float q = 0.f;
#pragma unroll
  for (int i = 0; i < 4; ++i) { const float d = v[i] - mean; q += d * d; }
#pragma unroll
  for (int off = 16; off > 0; off >>= 1) q += __shfl_xor(q, off, 32);
  const float rstd = rsqrtf(q * (1.0f / EDIM) + 1e-5f);
#pragma unroll
  for (int i = 0; i < 4; ++i) {
    const int c = lane + 32 * i;
    const float xh = (v[i] - mean) * rstd;
    xnf[(size_t)token * EDIM + c] = f2bf(xh * g[c] + bta[c]);
    xnb[(size_t)token * EDIM + c] = f2bf(xh * g[EDIM + c] + bta[EDIM + c]);
  }
}

// ---------------------------------------------------------------------------
// bf16 WMMA GEMM:
//   out[m, colOff+n] = bf16( A(M,K)bf16 @ Bt(N,K)bf16^T + bias[n] (+res[m,n]) )
// block = 128 threads (4 waves), tile 128(M) x 64(N), K step 32.
// Async double-buffered LDS staging; 8 back-to-back WMMAs per wave per step.
// ---------------------------------------------------------------------------
__global__ void __launch_bounds__(128)
gemm_bf16(const unsigned short* __restrict__ A, int lda,
          const unsigned short* __restrict__ Bt, int K,
          const float* __restrict__ bias,
          const unsigned short* __restrict__ res, int ldr,
          unsigned short* __restrict__ out, int ldc, int colOff)
{
  __shared__ unsigned short sA[2][128 * 32];  // row-major 128 rows x 32 k
  __shared__ unsigned short sB[2][64 * 32];   // row = output col, 32 k contiguous

  const int tid   = threadIdx.x;
  const int lane  = tid & 31;
  const int wave  = tid >> 5;
  const int mBase = blockIdx.x * 128;
  const int nBase = blockIdx.y * 64;

  v8f acc[2][4] = {};

  const int r  = tid >> 1;         // 0..63
  const int c0 = (tid & 1) * 16;   // 0 or 16

  // per-thread async staging of one K-step tile into buffer `buf`
  auto stage = [&](int k0, int buf) {
    const unsigned short* ga0 = A  + (size_t)(mBase + r)      * lda + k0 + c0;
    const unsigned short* ga1 = A  + (size_t)(mBase + r + 64) * lda + k0 + c0;
    const unsigned short* gb  = Bt + (size_t)(nBase + r)      * K   + k0 + c0;
    async_copy_b128(&sA[buf][r * 32 + c0],            ga0);
    async_copy_b128(&sA[buf][r * 32 + c0 + 8],        ga0 + 8);
    async_copy_b128(&sA[buf][(r + 64) * 32 + c0],     ga1);
    async_copy_b128(&sA[buf][(r + 64) * 32 + c0 + 8], ga1 + 8);
    async_copy_b128(&sB[buf][r * 32 + c0],            gb);
    async_copy_b128(&sB[buf][r * 32 + c0 + 8],        gb + 8);
  };

  const int nk = K / 32;
  stage(0, 0);

  for (int i = 0; i < nk; ++i) {
    const int buf = i & 1;
    wait_async0();       // own asyncs (tile i) done
    __syncthreads();     // -> everyone's tile-i writes visible; tile i-1 reads retired
    if (i + 1 < nk) stage(32 * (i + 1), buf ^ 1);

    // Load ALL fragments, then 8 independent WMMAs back-to-back.
    // A layout (16-bit 16x32): lane<16 -> K {0..7,16..23}, lane>=16 -> K {8..15,24..31}
    // B layout: lane holds col N=lane&15, K {0..7,16..23} (lane<16) / {8..15,24..31} (lane>=16)
    union F { v16bf v; uint4 q[2]; };
    F aF[2], bF[4];
    const int kbA = (lane < 16) ? 0 : 8;
    const int kbB = (lane < 16) ? 0 : 8;
#pragma unroll
    for (int s = 0; s < 2; ++s) {
      const int mA = wave * 32 + s * 16 + (lane & 15);
      aF[s].q[0] = *(const uint4*)&sA[buf][mA * 32 + kbA];
      aF[s].q[1] = *(const uint4*)&sA[buf][mA * 32 + kbA + 16];
    }
#pragma unroll
    for (int nt = 0; nt < 4; ++nt) {
      const int nB = nt * 16 + (lane & 15);
      bF[nt].q[0] = *(const uint4*)&sB[buf][nB * 32 + kbB];
      bF[nt].q[1] = *(const uint4*)&sB[buf][nB * 32 + kbB + 16];
    }
#pragma unroll
    for (int s = 0; s < 2; ++s)
#pragma unroll
      for (int nt = 0; nt < 4; ++nt)
        acc[s][nt] = __builtin_amdgcn_wmma_f32_16x16x32_bf16(
            false, aF[s].v, false, bF[nt].v, (short)0, acc[s][nt], false, false);
  }

  // epilogue: C/D layout -> VGPR v holds M = v + 8*(lane>=16), N = lane&15
#pragma unroll
  for (int s = 0; s < 2; ++s) {
    const int mRowBase = mBase + wave * 32 + s * 16 + ((lane < 16) ? 0 : 8);
#pragma unroll
    for (int nt = 0; nt < 4; ++nt) {
      const int n  = nBase + nt * 16 + (lane & 15);
      const float bv = bias[n];
#pragma unroll
      for (int v = 0; v < 8; ++v) {
        const int m = mRowBase + v;
        float val = acc[s][nt][v] + bv;
        if (res) val += bf2f(res[(size_t)m * ldr + n]);
        out[(size_t)m * ldc + colOff + n] = f2bf(val);
      }
    }
  }
}

// ---------------------------------------------------------------------------
// Mamba scan: softplus(delta), s = sum_j exp(delta*A[e,j]),
// forward inclusive cumsum + backward inclusive suffix-sum of xn*Bm,
// out = h*s + xn*D.  One thread per (batch, channel).
// ---------------------------------------------------------------------------
__global__ void __launch_bounds__(256)
scan_kernel(const unsigned short* __restrict__ xnf, const unsigned short* __restrict__ xnb,
            const unsigned short* __restrict__ xpf, const unsigned short* __restrict__ xpb,
            const float* __restrict__ Al, const float* __restrict__ Dl,
            unsigned short* __restrict__ outf, unsigned short* __restrict__ outb)
{
  const int idx = blockIdx.x * 256 + threadIdx.x;
  if (idx >= BATCHN * EDIM) return;
  const int b = idx >> 7, e = idx & 127;

  float Af[STDIM], Ab[STDIM];
#pragma unroll
  for (int j = 0; j < STDIM; ++j) {
    Af[j] = Al[(size_t)e * STDIM + j];
    Ab[j] = Al[(size_t)(EDIM + e) * STDIM + j];
  }
  const float Df = Dl[e], Db = Dl[EDIM + e];

  float acc = 0.f;
  for (int t = 0; t < SEQLEN; ++t) {
    const size_t row = (size_t)b * SEQLEN + t;
    const float xn = bf2f(xnf[row * EDIM + e]);
    const float xd = bf2f(xpf[row * 256 + e]);
    const float Bm = bf2f(xpf[row * 256 + EDIM + e]);
    const float dl = (xd > 20.f) ? xd : __logf(1.f + __expf(xd));
    float s = 0.f;
#pragma unroll
    for (int j = 0; j < STDIM; ++j) s += __expf(dl * Af[j]);
    acc += xn * Bm;
    outf[row * EDIM + e] = f2bf(acc * s + xn * Df);
  }

  acc = 0.f;
  for (int t = SEQLEN - 1; t >= 0; --t) {
    const size_t row = (size_t)b * SEQLEN + t;
    const float xn = bf2f(xnb[row * EDIM + e]);
    const float xd = bf2f(xpb[row * 256 + e]);
    const float Bm = bf2f(xpb[row * 256 + EDIM + e]);
    const float dl = (xd > 20.f) ? xd : __logf(1.f + __expf(xd));
    float s = 0.f;
#pragma unroll
    for (int j = 0; j < STDIM; ++j) s += __expf(dl * Ab[j]);
    acc += xn * Bm;
    outb[row * EDIM + e] = f2bf(acc * s + xn * Db);
  }
}

// ---------------------------------------------------------------------------
// Head MLP: 128 -> relu 128 -> relu 64 -> 1, one block per batch row
// ---------------------------------------------------------------------------
__global__ void __launch_bounds__(128)
head_kernel(const unsigned short* __restrict__ seq,
            const float* __restrict__ w1, const float* __restrict__ b1,
            const float* __restrict__ w2, const float* __restrict__ b2,
            const float* __restrict__ w3, const float* __restrict__ b3,
            float* __restrict__ out)
{
  __shared__ float h0[128], h1[128], h2[64];
  const int b = blockIdx.x, t = threadIdx.x;
  h0[t] = bf2f(seq[(size_t)b * SEQLEN * EDIM + t]);
  __syncthreads();
  float a = b1[t];
  for (int k = 0; k < 128; ++k) a += h0[k] * w1[k * 128 + t];
  h1[t] = fmaxf(a, 0.f);
  __syncthreads();
  if (t < 64) {
    float a2 = b2[t];
    for (int k = 0; k < 128; ++k) a2 += h1[k] * w2[k * 64 + t];
    h2[t] = fmaxf(a2, 0.f);
  }
  __syncthreads();
  if (t == 0) {
    float a3 = b3[0];
    for (int k = 0; k < 64; ++k) a3 += h2[k] * w3[k];
    out[b] = a3;
  }
}

// ---------------------------------------------------------------------------
extern "C" void kernel_launch(void* const* d_in, const int* in_sizes, int n_in,
                              void* d_out, int out_size, void* d_ws, size_t ws_size,
                              hipStream_t stream)
{
  const int*   x    = (const int*)  d_in[0];
  const float* emb  = (const float*)d_in[1];
  const float* cls  = (const float*)d_in[2];
  const float* ln_g = (const float*)d_in[3];
  const float* ln_b = (const float*)d_in[4];
  const float* wx   = (const float*)d_in[5];
  const float* bx   = (const float*)d_in[6];
  const float* Aw   = (const float*)d_in[7];
  const float* Dw   = (const float*)d_in[8];
  const float* wo   = (const float*)d_in[9];
  const float* bo   = (const float*)d_in[10];
  const float* mw   = (const float*)d_in[11];
  const float* mb   = (const float*)d_in[12];
  const float* w1   = (const float*)d_in[13];
  const float* b1   = (const float*)d_in[14];
  const float* w2   = (const float*)d_in[15];
  const float* b2   = (const float*)d_in[16];
  const float* w3   = (const float*)d_in[17];
  const float* b3   = (const float*)d_in[18];
  float* out = (float*)d_out;

  // bf16 workspace layout: 12*S activation elements (~252 MB) + ~1 MB weights
  const size_t S = (size_t)BATCHN * SEQLEN * EDIM;   // 10,485,760
  unsigned short* wsb = (unsigned short*)d_ws;
  unsigned short* seqBuf[2] = { wsb, wsb + S };
  unsigned short* xnf  = wsb + 2 * S;
  unsigned short* xnb  = wsb + 3 * S;
  unsigned short* xpf  = wsb + 4 * S;   // 2*S (B*SEQ x 256)
  unsigned short* xpb  = wsb + 6 * S;   // 2*S
  unsigned short* ocat = wsb + 8 * S;   // 2*S (f | bk concatenated)
  unsigned short* outf = wsb + 10 * S;
  unsigned short* outb = wsb + 11 * S;
  unsigned short* wxT  = wsb + 12 * S;                              // (L,2,256,128)
  unsigned short* woT  = wxT + (size_t)NLAYER * 2 * 256 * EDIM;     // (L,2,128,128)
  unsigned short* mwT  = woT + (size_t)NLAYER * 2 * EDIM * EDIM;    // (L,128,256)

  // one-shot weight transpose+convert to bf16 (N,K)
  for (int l = 0; l < NLAYER; ++l) {
    for (int d = 0; d < 2; ++d) {
      wprep_kernel<<<(EDIM * 256) / 256, 256, 0, stream>>>(
          wx + ((size_t)l * 2 + d) * EDIM * 256,
          wxT + ((size_t)l * 2 + d) * 256 * EDIM, EDIM, 256);
      wprep_kernel<<<(EDIM * EDIM) / 256, 256, 0, stream>>>(
          wo + ((size_t)l * 2 + d) * EDIM * EDIM,
          woT + ((size_t)l * 2 + d) * EDIM * EDIM, EDIM, EDIM);
    }
    wprep_kernel<<<(256 * EDIM) / 256, 256, 0, stream>>>(
        mw + (size_t)l * 256 * EDIM, mwT + (size_t)l * EDIM * 256, 256, EDIM);
  }

  embed_kernel<<<(unsigned)((S + 255) / 256), 256, 0, stream>>>(x, emb, cls, seqBuf[0]);

  const int M = BATCHN * SEQLEN;  // 81920, multiple of 128
  for (int l = 0; l < NLAYER; ++l) {
    const unsigned short* in  = seqBuf[l & 1];
    unsigned short*       nxt = seqBuf[(l + 1) & 1];

    ln_kernel<<<M / 8, 256, 0, stream>>>(in, ln_g + (size_t)l * 2 * EDIM,
                                         ln_b + (size_t)l * 2 * EDIM, xnf, xnb);

    // xp = xn @ wx + bx  (K=128, N=256), both directions
    {
      dim3 g(M / 128, 256 / 64);
      gemm_bf16<<<g, 128, 0, stream>>>(xnf, EDIM, wxT + (size_t)l * 2 * 256 * EDIM, EDIM,
                                       bx + (size_t)l * 2 * 256, nullptr, 0, xpf, 256, 0);
      gemm_bf16<<<g, 128, 0, stream>>>(xnb, EDIM, wxT + ((size_t)l * 2 + 1) * 256 * EDIM, EDIM,
                                       bx + ((size_t)l * 2 + 1) * 256, nullptr, 0, xpb, 256, 0);
    }

    scan_kernel<<<(BATCHN * EDIM) / 256, 256, 0, stream>>>(
        xnf, xnb, xpf, xpb,
        Aw + (size_t)l * 2 * EDIM * STDIM, Dw + (size_t)l * 2 * EDIM, outf, outb);

    // f / bk = scan_out @ wo + bo + residual(in)  (K=128, N=128) -> ocat cols
    {
      dim3 g(M / 128, 128 / 64);
      gemm_bf16<<<g, 128, 0, stream>>>(outf, EDIM, woT + (size_t)l * 2 * EDIM * EDIM, EDIM,
                                       bo + (size_t)l * 2 * EDIM, in, EDIM, ocat, 256, 0);
      gemm_bf16<<<g, 128, 0, stream>>>(outb, EDIM, woT + ((size_t)l * 2 + 1) * EDIM * EDIM, EDIM,
                                       bo + ((size_t)l * 2 + 1) * EDIM, in, EDIM, ocat, 256, 128);
    }

    // nxt = [f | bk] @ mw + mb  (K=256, N=128)
    {
      dim3 g(M / 128, 128 / 64);
      gemm_bf16<<<g, 128, 0, stream>>>(ocat, 256, mwT + (size_t)l * EDIM * 256, 256,
                                       mb + (size_t)l * EDIM, nullptr, 0, nxt, EDIM, 0);
    }
  }

  head_kernel<<<BATCHN, 128, 0, stream>>>(seqBuf[0], w1, b1, w2, b2, w3, b3, out);

  (void)in_sizes; (void)n_in; (void)out_size; (void)ws_size;
}
